// MoELayer_30356828848665
// MI455X (gfx1250) — matrix operand order repaired and checked by reading, not compiled, not run on hardware
//
#include <hip/hip_runtime.h>
#include <hip/hip_bf16.h>
#include <math.h>

// Problem constants (B,S,H,F,E,K) = (2,1024,2048,5504,8,2)
#define T_TOKENS 2048
#define H_DIM    2048
#define F_DIM    5504
#define E_NUM    8

typedef __attribute__((ext_vector_type(16))) __bf16 v16bf;
typedef __attribute__((ext_vector_type(8)))  float  v8f;
typedef __attribute__((ext_vector_type(4)))  unsigned int u32x4;
typedef __attribute__((ext_vector_type(8)))  int          i32x8;
typedef __attribute__((ext_vector_type(4)))  int          i32x4;

#if defined(__gfx1250__) && __has_builtin(__builtin_amdgcn_tensor_load_to_lds)
#define USE_TDM 1
#else
#define USE_TDM 0
#endif

// Scheduling hints: pin DS-read group ahead of WMMA group per K-step so the
// compiler emits partial s_wait_dscnt instead of dscnt==0 before every WMMA.
#if defined(__gfx1250__) && __has_builtin(__builtin_amdgcn_sched_group_barrier)
#define SCHED_PIPELINE(nloads, nmma)                              \
  do {                                                            \
    __builtin_amdgcn_sched_group_barrier(0x100, (nloads), 0);     \
    __builtin_amdgcn_sched_group_barrier(0x008, (nmma), 0);       \
  } while (0)
#else
#define SCHED_PIPELINE(nloads, nmma) do {} while (0)
#endif

// ---------------------------------------------------------------------------
// Workspace layout (bytes). Requires ws_size >= ~585 MiB.
//   small control data in [0, 192KB), then:
//   P   @  4 MiB : bf16[4096+64][F]  compact silu(g)*u rows (~43.7 MiB)
//   xb  @ 52 MiB : bf16 x            (8 MiB)
//   gwb @ 62 MiB : bf16 gate_w       (172 MiB)
//   uwb @236 MiB : bf16 up_w         (172 MiB)
//   dwb @410 MiB : bf16 down_w       (172 MiB)
// ---------------------------------------------------------------------------
#define WS_COUNTS   0
#define WS_OFFSETS  32
#define WS_TOKLIST  64
#define WS_WLIST    (WS_TOKLIST + E_NUM * T_TOKENS * 4)
#define WS_EIDS     (WS_WLIST + E_NUM * T_TOKENS * 4)
#define WS_EPROBS   (WS_EIDS + T_TOKENS * 8)
#define WS_P        ((size_t)4 << 20)
#define WS_XB       ((size_t)52 << 20)
#define WS_GWB      ((size_t)62 << 20)
#define WS_UWB      ((size_t)236 << 20)
#define WS_DWB      ((size_t)410 << 20)

// ----------------------- WMMA fragment loaders (bf16) ----------------------
// A-matrix 16x32 bf16: lane half h=(lane>>4): elems 0..15 -> K=h*16+i
// (same K order as B). Row m = lane&15.
__device__ __forceinline__ v16bf load_a_frag(const __bf16* row, int half) {
  v16bf f;
  const __bf16* p0 = row + half * 16;
  const __bf16* p1 = row + half * 16 + 8;
#pragma unroll
  for (int i = 0; i < 8; ++i) f[i] = p0[i];
#pragma unroll
  for (int i = 0; i < 8; ++i) f[i + 8] = p1[i];
  return f;
}
// B-matrix 32x16 bf16: lanes 0-15 hold K=0..15 (2/VGPR), lanes 16-31 K=16..31;
// col n = lane&15. W tiles in LDS are row-major [n][k] -> 16 contiguous bf16.
__device__ __forceinline__ v16bf load_b_frag(const __bf16* row, int half) {
  v16bf f;
  const __bf16* p = row + half * 16;
#pragma unroll
  for (int i = 0; i < 16; ++i) f[i] = p[i];
  return f;
}

// --------------------------- TDM 2D tile loader ----------------------------
// Loads a [rows x 64] bf16 tile (row stride = stride_elems in memory) into LDS
// with pad: 128B of data then 16B pad -> LDS row stride 144B (72 bf16), which
// matches the shared-tile layout used by the fragment loaders.
__device__ __forceinline__ void tdm_load_tile64(unsigned lds_addr, const void* gptr,
                                                int rows, long long stride_elems) {
#if USE_TDM
  unsigned long long ga = (unsigned long long)(uintptr_t)gptr;
  u32x4 g0;
  g0[0] = 1u;                                            // count=1, user mode
  g0[1] = lds_addr;                                      // LDS byte address
  g0[2] = (unsigned)(ga & 0xFFFFFFFFu);                  // global_addr[31:0]
  g0[3] = (unsigned)((ga >> 32) & 0x01FFFFFFu) | (2u << 30);  // addr[56:32]|type=2
  i32x8 g1;
  // data_size=1 (2B), pad_enable=1, pad_interval=4 (32 DW = 128B), pad_amount=3 (4 DW = 16B)
  g1[0] = (int)((1u << 16) | (1u << 20) | (4u << 22) | (3u << 25));
  unsigned dim0 = 0x7FFFFFFFu, dim1 = 0x7FFFFFFFu;       // huge tensor dims: no OOB clip
  g1[1] = (int)((dim0 & 0xFFFFu) << 16);                 // [15:0] atomic addr=0, [31:16] dim0.lo
  g1[2] = (int)((dim0 >> 16) | ((dim1 & 0xFFFFu) << 16));
  g1[3] = (int)((dim1 >> 16) | (64u << 16));             // tile_dim0 = 64 elems
  g1[4] = (int)((unsigned)rows & 0xFFFFu);               // tile_dim1 = rows, tile_dim2 = 0
  unsigned long long s0 = (unsigned long long)stride_elems;
  g1[5] = (int)(s0 & 0xFFFFFFFFull);                     // tensor_dim0_stride[31:0]
  g1[6] = (int)((s0 >> 32) & 0xFFFFull);                 // [47:32]; dim1_stride = 0
  g1[7] = 0;
  i32x4 z4 = {0, 0, 0, 0};
#if __clang_major__ >= 23
  i32x8 z8 = {0, 0, 0, 0, 0, 0, 0, 0};
  __builtin_amdgcn_tensor_load_to_lds(g0, g1, z4, z4, z8, 0);
#else
  __builtin_amdgcn_tensor_load_to_lds(g0, g1, z4, z4, 0);
#endif
#else
  (void)lds_addr; (void)gptr; (void)rows; (void)stride_elems;
#endif
}
__device__ __forceinline__ void tdm_wait_all() {
#if USE_TDM
  __builtin_amdgcn_s_wait_tensorcnt(0);
#endif
}

// --------------------------------- zero ------------------------------------
__global__ __launch_bounds__(256) void zero_kernel(float4* __restrict__ out4) {
  size_t i = (size_t)blockIdx.x * blockDim.x + threadIdx.x;
  out4[i] = make_float4(0.f, 0.f, 0.f, 0.f);   // grid sized exactly T*H/4
}

// --------------------------- fp32 -> bf16 convert ---------------------------
__global__ __launch_bounds__(256) void cvt_bf16_kernel(const float4* __restrict__ src,
                                                       uint4* __restrict__ dst,
                                                       long long n8) {
  long long i = (long long)blockIdx.x * blockDim.x + threadIdx.x;
  const long long stride = (long long)gridDim.x * blockDim.x;
  for (; i < n8; i += stride) {
    float4 a = src[2 * i], b = src[2 * i + 1];
    union { __bf16 h[8]; uint4 u; } t;
    t.h[0] = (__bf16)a.x; t.h[1] = (__bf16)a.y; t.h[2] = (__bf16)a.z; t.h[3] = (__bf16)a.w;
    t.h[4] = (__bf16)b.x; t.h[5] = (__bf16)b.y; t.h[6] = (__bf16)b.z; t.h[7] = (__bf16)b.w;
    dst[i] = t.u;
  }
}

// -------------------------------- router -----------------------------------
__global__ __launch_bounds__(256) void router_kernel(
    const float* __restrict__ x, const float* __restrict__ rw,
    int2* __restrict__ eids, float2* __restrict__ eprobs) {
  const int t = blockIdx.x;
  const int tid = threadIdx.x;
  const float* xr = x + (size_t)t * H_DIM;

  float acc[E_NUM] = {0.f, 0.f, 0.f, 0.f, 0.f, 0.f, 0.f, 0.f};
  for (int h = tid; h < H_DIM; h += 256) {
    float xv = xr[h];
#pragma unroll
    for (int e = 0; e < E_NUM; ++e) acc[e] += xv * rw[e * H_DIM + h];
  }
  __shared__ float part[E_NUM][256];
#pragma unroll
  for (int e = 0; e < E_NUM; ++e) part[e][tid] = acc[e];
  __syncthreads();
  for (int s = 128; s > 0; s >>= 1) {
    if (tid < s) {
#pragma unroll
      for (int e = 0; e < E_NUM; ++e) part[e][tid] += part[e][tid + s];
    }
    __syncthreads();
  }
  if (tid == 0) {
    float l[E_NUM];
#pragma unroll
    for (int e = 0; e < E_NUM; ++e) l[e] = part[e][0];
    int i1 = 0;
    for (int e = 1; e < E_NUM; ++e) if (l[e] > l[i1]) i1 = e;
    int i2 = (i1 == 0) ? 1 : 0;
    for (int e = 0; e < E_NUM; ++e) if (e != i1 && l[e] > l[i2]) i2 = e;
    float m = fmaxf(l[i1], l[i2]);
    float p1 = __expf(l[i1] - m), p2 = __expf(l[i2] - m);
    float inv = 1.f / (p1 + p2);
    eids[t] = make_int2(i1, i2);
    eprobs[t] = make_float2(p1 * inv, p2 * inv);
  }
}

// ------------------------------ build lists --------------------------------
// Single block, 8 waves: wave e does a stable ballot-compaction for expert e.
__global__ __launch_bounds__(256) void build_lists_kernel(
    const int2* __restrict__ eids, const float2* __restrict__ eprobs,
    int* __restrict__ counts, int* __restrict__ offsets,
    int* __restrict__ tokenlist, float* __restrict__ weightlist) {
  const int lane = threadIdx.x & 31;
  const int e = threadIdx.x >> 5;
  __shared__ int sh_cnt[E_NUM];

  int run = 0;
  for (int base = 0; base < T_TOKENS; base += 32) {
    int t = base + lane;
    int2 id = eids[t];
    float2 pr = eprobs[t];
    bool mx = (id.x == e), my = (id.y == e);
    bool m = mx || my;
    float w = mx ? pr.x : pr.y;
    unsigned long long mask = __ballot(m);
    if (m) {
      int pos = run + __popcll(mask & ((1ull << lane) - 1ull));
      tokenlist[e * T_TOKENS + pos] = t;
      weightlist[e * T_TOKENS + pos] = w;
    }
    run += __popcll(mask);
  }
  if (lane == 0) sh_cnt[e] = run;
  __syncthreads();
  if (threadIdx.x == 0) {
    int s = 0;
    for (int i = 0; i < E_NUM; ++i) { counts[i] = sh_cnt[i]; offsets[i] = s; s += sh_cnt[i]; }
  }
}

// ------------------------- expert MLP stage 1 ------------------------------
// Block tile: 64 tokens x 128 F, gate and up fused. Double-buffered LDS;
// weight tiles via TDM (waves 0/1), X tile VALU-gathered from bf16 xb.
__global__ __launch_bounds__(256) void moe_mlp1_kernel(
    const __bf16* __restrict__ xb, const __bf16* __restrict__ gwb,
    const __bf16* __restrict__ uwb, const int* __restrict__ counts,
    const int* __restrict__ offsets, const int* __restrict__ tokenlist,
    __bf16* __restrict__ P) {
  const int e = blockIdx.y;
  const int Ne = counts[e];
  const int tileM = blockIdx.z;
  if (tileM * 64 >= Ne) return;
  const int fBase = blockIdx.x * 128;
  const int expOff = offsets[e];

  __shared__ __bf16 Xs[2][64 * 72];
  __shared__ __bf16 Wgs[2][128 * 72];
  __shared__ __bf16 Wus[2][128 * 72];
  __shared__ int toks[64];

  const int tid = threadIdx.x;
  if (tid < 64) {
    int rl = tileM * 64 + tid;
    toks[tid] = (rl < Ne) ? tokenlist[e * T_TOKENS + rl] : 0;  // clamp; rows discarded later
  }
  __syncthreads();

  const int lane = tid & 31;
  const int wid = tid >> 5;   // 8 waves
  const int tm = wid >> 1;    // token sub-tile 0..3
  const int fn = wid & 1;     // F half (64 cols each)
  const int mrow = lane & 15;
  const int half = lane >> 4;

  const int xr = tid >> 2, xc = (tid & 3) * 16;   // X staging: 32 B/thread
  const __bf16* gbase = gwb + ((size_t)e * F_DIM + fBase) * H_DIM;
  const __bf16* ubase = uwb + ((size_t)e * F_DIM + fBase) * H_DIM;

  v8f cg[4] = {};
  v8f cu[4] = {};

  auto stage = [&](int b, int k0) {
    // X tile (gathered rows, contiguous 32B per thread)
    {
      const uint4* src = (const uint4*)(xb + (size_t)toks[xr] * H_DIM + k0 + xc);
      uint4* dst = (uint4*)&Xs[b][xr * 72 + xc];
      dst[0] = src[0];
      dst[1] = src[1];
    }
#if USE_TDM
    if (wid == 0)
      tdm_load_tile64((unsigned)(uintptr_t)&Wgs[b][0], gbase + k0, 128, H_DIM);
    else if (wid == 1)
      tdm_load_tile64((unsigned)(uintptr_t)&Wus[b][0], ubase + k0, 128, H_DIM);
#else
    const int wr = tid >> 1, wc = (tid & 1) * 32;   // 64 B/thread per tile
    {
      const uint4* sg = (const uint4*)(gbase + (size_t)wr * H_DIM + k0 + wc);
      const uint4* su = (const uint4*)(ubase + (size_t)wr * H_DIM + k0 + wc);
      uint4* dg = (uint4*)&Wgs[b][wr * 72 + wc];
      uint4* du = (uint4*)&Wus[b][wr * 72 + wc];
#pragma unroll
      for (int q = 0; q < 4; ++q) { dg[q] = sg[q]; du[q] = su[q]; }
    }
#endif
  };

  stage(0, 0);  // prologue
  int buf = 0;
  for (int k0 = 0; k0 < H_DIM; k0 += 64) {
    tdm_wait_all();        // current buffer's DMA complete (issuing waves)
    __syncthreads();       // DMA + staging visible; prev compute done
    const int nxt = buf ^ 1;
    if (k0 + 64 < H_DIM) stage(nxt, k0 + 64);  // overlap next DMA/stage with compute
#pragma unroll
    for (int kk = 0; kk < 64; kk += 32) {
      v16bf a = load_a_frag(&Xs[buf][(tm * 16 + mrow) * 72 + kk], half);
      v16bf bg[4], bu[4];
#pragma unroll
      for (int j = 0; j < 4; ++j) {
        bg[j] = load_b_frag(&Wgs[buf][(fn * 64 + j * 16 + mrow) * 72 + kk], half);
        bu[j] = load_b_frag(&Wus[buf][(fn * 64 + j * 16 + mrow) * 72 + kk], half);
      }
#pragma unroll
      for (int j = 0; j < 4; ++j) {
        cg[j] = __builtin_amdgcn_wmma_f32_16x16x32_bf16(
            false, a, false, bg[j], (short)0, cg[j], false, false);
        cu[j] = __builtin_amdgcn_wmma_f32_16x16x32_bf16(
            false, a, false, bu[j], (short)0, cu[j], false, false);
      }
      // 18 ds_load_b128 (a:2, 8 B-frags:16) first, then 8 WMMAs -> partial waits
      SCHED_PIPELINE(18, 8);
    }
    buf = nxt;
  }

  // epilogue: p = silu(g)*u -> compact P rows (bf16)
  const int n = lane & 15;
  const int rsub = tileM * 64 + tm * 16;
#pragma unroll
  for (int j = 0; j < 4; ++j) {
#pragma unroll
    for (int v = 0; v < 8; ++v) {
      int rl = rsub + v + 8 * half;
      if (rl < Ne) {
        float g = cg[j][v], u = cu[j][v];
        float pv = (g / (1.f + __expf(-g))) * u;
        P[(size_t)(expOff + rl) * F_DIM + (fBase + fn * 64 + j * 16 + n)] = (__bf16)pv;
      }
    }
  }
}

// ------------------------- expert MLP stage 2 ------------------------------
// Y[64 x 128] = P_rows @ down^T over F; both tiles are contiguous 2D -> TDM.
__global__ __launch_bounds__(256) void moe_mlp2_kernel(
    const __bf16* __restrict__ P, const __bf16* __restrict__ dwb,
    const int* __restrict__ counts, const int* __restrict__ offsets,
    const int* __restrict__ tokenlist, const float* __restrict__ weightlist,
    float* __restrict__ out) {
  const int e = blockIdx.y;
  const int Ne = counts[e];
  const int tileM = blockIdx.z;
  if (tileM * 64 >= Ne) return;
  const int hBase = blockIdx.x * 128;
  const int expOff = offsets[e];

  __shared__ __bf16 Ps[2][64 * 72];
  __shared__ __bf16 Wds[2][128 * 72];

  const int tid = threadIdx.x;
  const int lane = tid & 31;
  const int wid = tid >> 5;
  const int tm = wid >> 1;
  const int hn = wid & 1;
  const int mrow = lane & 15;
  const int half = lane >> 4;

  const __bf16* pbase = P + (size_t)(expOff + tileM * 64) * F_DIM;
  const __bf16* dbase = dwb + ((size_t)e * H_DIM + hBase) * F_DIM;

  v8f cy[4] = {};

  auto stage = [&](int b, int k0) {
#if USE_TDM
    if (wid == 0)
      tdm_load_tile64((unsigned)(uintptr_t)&Wds[b][0], dbase + k0, 128, F_DIM);
    else if (wid == 1)
      tdm_load_tile64((unsigned)(uintptr_t)&Ps[b][0], pbase + k0, 64, F_DIM);
#else
    {
      const int wr = tid >> 1, wc = (tid & 1) * 32;
      const uint4* sd = (const uint4*)(dbase + (size_t)wr * F_DIM + k0 + wc);
      uint4* dd = (uint4*)&Wds[b][wr * 72 + wc];
#pragma unroll
      for (int q = 0; q < 4; ++q) dd[q] = sd[q];
    }
    {
      const int pr = tid >> 2, pc = (tid & 3) * 16;
      const uint4* sp = (const uint4*)(pbase + (size_t)pr * F_DIM + k0 + pc);
      uint4* dp = (uint4*)&Ps[b][pr * 72 + pc];
      dp[0] = sp[0];
      dp[1] = sp[1];
    }
#endif
  };

  stage(0, 0);
  int buf = 0;
  for (int k0 = 0; k0 < F_DIM; k0 += 64) {
    tdm_wait_all();
    __syncthreads();
    const int nxt = buf ^ 1;
    if (k0 + 64 < F_DIM) stage(nxt, k0 + 64);
#pragma unroll
    for (int kk = 0; kk < 64; kk += 32) {
      v16bf a = load_a_frag(&Ps[buf][(tm * 16 + mrow) * 72 + kk], half);
      v16bf bd[4];
#pragma unroll
      for (int j = 0; j < 4; ++j)
        bd[j] = load_b_frag(&Wds[buf][(hn * 64 + j * 16 + mrow) * 72 + kk], half);
#pragma unroll
      for (int j = 0; j < 4; ++j)
        cy[j] = __builtin_amdgcn_wmma_f32_16x16x32_bf16(
            false, a, false, bd[j], (short)0, cy[j], false, false);
      // 10 ds_load_b128 (a:2, 4 B-frags:8) first, then 4 WMMAs
      SCHED_PIPELINE(10, 4);
    }
    buf = nxt;
  }

  // epilogue: out[t, h] += combine_weight * y
  const int n = lane & 15;
  const int rsub = tileM * 64 + tm * 16;
  int tok[8];
  float cw[8];
#pragma unroll
  for (int v = 0; v < 8; ++v) {
    int rl = rsub + v + 8 * half;
    if (rl < Ne) {
      tok[v] = tokenlist[e * T_TOKENS + rl];
      cw[v] = weightlist[e * T_TOKENS + rl];
    } else {
      tok[v] = -1;
      cw[v] = 0.f;
    }
  }
#pragma unroll
  for (int j = 0; j < 4; ++j) {
#pragma unroll
    for (int v = 0; v < 8; ++v) {
      if (tok[v] >= 0) {
        atomicAdd(out + (size_t)tok[v] * H_DIM + (hBase + hn * 64 + j * 16 + n),
                  cw[v] * cy[j][v]);
      }
    }
  }
}

// --------------------------------- launch ----------------------------------
extern "C" void kernel_launch(void* const* d_in, const int* in_sizes, int n_in,
                              void* d_out, int out_size, void* d_ws, size_t ws_size,
                              hipStream_t stream) {
  (void)in_sizes; (void)n_in; (void)out_size; (void)ws_size;
  const float* x        = (const float*)d_in[0];
  const float* router_w = (const float*)d_in[1];
  const float* gate_w   = (const float*)d_in[2];
  const float* up_w     = (const float*)d_in[3];
  const float* down_w   = (const float*)d_in[4];
  float* out = (float*)d_out;

  char* ws = (char*)d_ws;
  int*    counts     = (int*)(ws + WS_COUNTS);
  int*    offsets    = (int*)(ws + WS_OFFSETS);
  int*    tokenlist  = (int*)(ws + WS_TOKLIST);
  float*  weightlist = (float*)(ws + WS_WLIST);
  int2*   eids       = (int2*)(ws + WS_EIDS);
  float2* eprobs     = (float2*)(ws + WS_EPROBS);
  __bf16* P          = (__bf16*)(ws + WS_P);
  __bf16* xb         = (__bf16*)(ws + WS_XB);
  __bf16* gwb        = (__bf16*)(ws + WS_GWB);
  __bf16* uwb        = (__bf16*)(ws + WS_UWB);
  __bf16* dwb        = (__bf16*)(ws + WS_DWB);

  const long long nW8 = (long long)E_NUM * F_DIM * H_DIM / 8;  // per weight tensor
  const long long nX8 = (long long)T_TOKENS * H_DIM / 8;

  // fp32 -> bf16 pre-conversion (one pass; bf16 working set then lives in L2)
  cvt_bf16_kernel<<<4096, 256, 0, stream>>>((const float4*)x, (uint4*)xb, nX8);
  cvt_bf16_kernel<<<4096, 256, 0, stream>>>((const float4*)gate_w, (uint4*)gwb, nW8);
  cvt_bf16_kernel<<<4096, 256, 0, stream>>>((const float4*)up_w, (uint4*)uwb, nW8);
  cvt_bf16_kernel<<<4096, 256, 0, stream>>>((const float4*)down_w, (uint4*)dwb, nW8);

  // zero output (atomicAdd target): T*H/4 float4 -> 4096 blocks x 256
  zero_kernel<<<(T_TOKENS * H_DIM) / (4 * 256), 256, 0, stream>>>((float4*)out);
  // router top-2 + softmax (fp32)
  router_kernel<<<T_TOKENS, 256, 0, stream>>>(x, router_w, eids, eprobs);
  // deterministic per-expert compaction
  build_lists_kernel<<<1, 256, 0, stream>>>(eids, eprobs, counts, offsets,
                                            tokenlist, weightlist);
  // gate/up + SwiGLU -> P
  moe_mlp1_kernel<<<dim3(F_DIM / 128, E_NUM, T_TOKENS / 64), 256, 0, stream>>>(
      xb, gwb, uwb, counts, offsets, tokenlist, P);
  // down proj, scale by combine weight, accumulate
  moe_mlp2_kernel<<<dim3(H_DIM / 128, E_NUM, T_TOKENS / 64), 256, 0, stream>>>(
      P, dwb, counts, offsets, tokenlist, weightlist, out);
}
